// VrtConv_50852412784796
// MI455X (gfx1250) — matrix operand order repaired and checked by reading, not compiled, not run on hardware
//
#include <hip/hip_runtime.h>
#include <hip/hip_bf16.h>
#include <stdint.h>

// CDNA5 / gfx1250 WMMA types
typedef __attribute__((ext_vector_type(16))) __bf16 v16bf;
typedef __attribute__((ext_vector_type(8)))  __bf16 v8bf;
typedef __attribute__((ext_vector_type(8)))  float  v8f;

#define S_LEN 256     // slide (GEMM M)
#define CH    128     // channels (GEMM N and contraction channel dim)
#define DW    128     // embedding width
#define NGEMM 33      // 18 (h=1) + 10 (h=2) + 5 (h=3)
#define KSPLIT 8      // K-chunks per GEMM (fast path)

union Frag16 { v16bf v; v8bf h[2]; };

__device__ inline v8bf cvt8(float4 x, float4 y) {
  v8bf r;
  r[0] = (__bf16)x.x; r[1] = (__bf16)x.y; r[2] = (__bf16)x.z; r[3] = (__bf16)x.w;
  r[4] = (__bf16)y.x; r[5] = (__bf16)y.y; r[6] = (__bf16)y.z; r[7] = (__bf16)y.w;
  return r;
}

// A fragment 16x32 bf16: lane<16 -> M=lane, K chunks {0..7,16..23};
// lane>=16 -> M=lane-16, K chunks {8..15,24..31}.  (STRIDE = LDS row stride, bf16)
template<int STRIDE>
__device__ inline v16bf load_frag_a(const __bf16* As, int r0, int kk, int lane) {
  const int row = r0 + (lane & 15);
  const int c0  = kk + (lane >> 4) * 8;
  const __bf16* base = As + row * STRIDE + c0;
  Frag16 f;
  f.h[0] = *(const v8bf*)(base);
  f.h[1] = *(const v8bf*)(base + 16);
  return f.v;
}

// B fragment 32x16 bf16 (KxN): lane<16 -> N=lane, K chunks {0..7,16..23}; lane>=16 -> N=lane-16, K chunks {8..15,24..31}.
template<int STRIDE>
__device__ inline v16bf load_frag_b(const __bf16* Bs, int n0, int kk, int lane) {
  const int col = n0 + (lane & 15);
  const int k0  = kk + (lane >> 4) * 8;
  const __bf16* base = Bs + col * STRIDE + k0;
  Frag16 f;
  f.h[0] = *(const v8bf*)(base);
  f.h[1] = *(const v8bf*)(base + 16);
  return f.v;
}

__device__ inline void decode_gemm(int g, int& h_idx, int& p, int& j) {
  if (g < 18)      { h_idx = 0; p = g / 6;        j = g % 6; }
  else if (g < 28) { h_idx = 1; p = (g - 18) / 5; j = (g - 18) % 5; }
  else             { h_idx = 2; p = 0;            j = g - 28; }
}

// ---------------- CDNA5 async global->LDS helpers ----------------
__device__ inline void async_b128(uint32_t lds_off, const void* gptr) {
  asm volatile("global_load_async_to_lds_b128 %0, %1, off"
               :: "v"(lds_off), "v"(gptr) : "memory");
}
__device__ inline void wait_async0() {
  asm volatile("s_wait_asynccnt 0" ::: "memory");
}

// ================= FAST PATH =================
// Prepass: fp32 -> bf16 (8 elements / thread, b128 in / b128 out)
__global__ __launch_bounds__(256)
void cvt_f32_bf16(const float* __restrict__ in, __bf16* __restrict__ out, int n8) {
  const int i = blockIdx.x * 256 + threadIdx.x;
  if (i >= n8) return;
  const float4* p = (const float4*)in + (size_t)i * 2;
  ((v8bf*)out)[i] = cvt8(p[0], p[1]);
}

#define F_MT 128
#define F_KB 64
#define F_STRIDE 72   // padded row stride in bf16 (144B) -> conflict-free b128 frag reads

// One K-chunk of one GEMM per block: partial C into part[(g*8+kc)][s][o]
__global__ __launch_bounds__(256)
void vrtconv_gemm_async(const __bf16* __restrict__ ecb,
                        const __bf16* __restrict__ w1b,
                        const __bf16* __restrict__ w2b,
                        const __bf16* __restrict__ w3b,
                        float* __restrict__ part) {
  __shared__ __bf16 As[2][F_MT][F_STRIDE];
  __shared__ __bf16 Bs[2][F_MT][F_STRIDE];

  const int g = blockIdx.z;
  int h_idx, p, j;
  decode_gemm(g, h_idx, p, j);
  const int h = h_idx + 1;
  const __bf16* __restrict__ w = (h_idx == 0) ? w1b : (h_idx == 1) ? w2b : w3b;

  const int mBase = blockIdx.x * F_MT;      // s tile (0 or 128)
  const int kc    = blockIdx.y;             // K chunk 0..7
  const int K     = h * CH * DW;            // 16384*h
  const int Kc    = K / KSPLIT;
  const int k0b   = kc * Kc;
  const int ns    = Kc / F_KB;              // 32*h stages

  const int tid  = threadIdx.x;
  const int lane = tid & 31;
  const int wv   = tid >> 5;                // 0..7
  const int wm   = (wv >> 1) * 32;          // wave M offset (0..96)
  const int wn   = (wv & 1)  * 64;          // wave N offset (0/64)

  const uint32_t asbase = (uint32_t)(uintptr_t)&As[0][0][0];
  const uint32_t bsbase = (uint32_t)(uintptr_t)&Bs[0][0][0];
  const uint32_t bufbytes = F_MT * F_STRIDE * 2;

  v8f acc[2][4];
#pragma unroll
  for (int a0 = 0; a0 < 2; ++a0)
#pragma unroll
    for (int b0 = 0; b0 < 4; ++b0)
#pragma unroll
      for (int e = 0; e < 8; ++e) acc[a0][b0][e] = 0.f;

  // async stage: 128 rows x 64 bf16 for A and B = 1024 b128 chunks each side / 256 thr
  auto issue_stage = [&](int st, int buf) {
    const int k0 = k0b + st * F_KB;
    const int ay = k0 >> 7;                 // (a*h + y); 64-block sits in one (a,y)
    const int d0 = k0 & 127;
    const int a  = ay / h;
    const int y  = ay % h;
#pragma unroll
    for (int t = 0; t < 4; ++t) {
      const int cid = t * 256 + tid;
      const int row = cid >> 3;             // 0..127
      const int col = (cid & 7) * 8;        // 0..56 (bf16)
      const uint32_t loff = buf * bufbytes + (uint32_t)(row * F_STRIDE + col) * 2;
      const __bf16* ga = ecb + ((((size_t)a * S_LEN + (mBase + row)) * 3
                                 + (size_t)(p + y)) << 7) + d0 + col;
      async_b128(asbase + loff, ga);
      const __bf16* gb = w + (size_t)(j * CH + row) * K + k0 + col;
      async_b128(bsbase + loff, gb);
    }
  };

  auto compute = [&](int buf) {
    const __bf16* ab = &As[buf][0][0];
    const __bf16* bb = &Bs[buf][0][0];
#pragma unroll
    for (int kk = 0; kk < F_KB; kk += 32) {
      v16bf af[2], bfr[4];
#pragma unroll
      for (int mt = 0; mt < 2; ++mt) af[mt]  = load_frag_a<F_STRIDE>(ab, wm + mt * 16, kk, lane);
#pragma unroll
      for (int nt = 0; nt < 4; ++nt) bfr[nt] = load_frag_b<F_STRIDE>(bb, wn + nt * 16, kk, lane);
#pragma unroll
      for (int mt = 0; mt < 2; ++mt)
#pragma unroll
        for (int nt = 0; nt < 4; ++nt)
          acc[mt][nt] = __builtin_amdgcn_wmma_f32_16x16x32_bf16(
              false, af[mt], false, bfr[nt], (short)0, acc[mt][nt], false, false);
    }
  };

  issue_stage(0, 0);
  wait_async0();
  __syncthreads();
  for (int st = 0; st < ns; ++st) {
    const int  cur  = st & 1;
    const bool more = (st + 1) < ns;
    if (more) issue_stage(st + 1, cur ^ 1);   // overlap async copy with WMMA
    compute(cur);
    if (more) wait_async0();
    __syncthreads();
  }

  float* pb = part + (size_t)(g * KSPLIT + kc) * (S_LEN * CH);
  const int mrow = (lane >> 4) * 8;
  const int ncol = lane & 15;
#pragma unroll
  for (int mt = 0; mt < 2; ++mt)
#pragma unroll
    for (int nt = 0; nt < 4; ++nt) {
      const int sOut = mBase + wm + mt * 16 + mrow;
      const int oOut = wn + nt * 16 + ncol;
#pragma unroll
      for (int r = 0; r < 8; ++r)
        pb[(size_t)(sOut + r) * CH + oOut] = acc[mt][nt][r];
    }
}

// sum K-chunks, max over positions, bias, ReLU, transpose to (B,S,1,16)
__global__ __launch_bounds__(256)
void vrtconv_reduce_split(const float* __restrict__ part,
                          const float* __restrict__ b1,
                          const float* __restrict__ b2,
                          const float* __restrict__ b3,
                          float* __restrict__ out) {
  const int t = blockIdx.x * blockDim.x + threadIdx.x;
  if (t >= CH * S_LEN * 16) return;
  const int kidx = t & 15;
  const int s    = (t >> 4) & (S_LEN - 1);
  const int o    = t >> 12;
  const int h_idx = kidx % 3;      // kidx = h_idx + 3*j
  const int j     = kidx / 3;
  const int P     = 3 - h_idx;
  const int gbase = (h_idx == 0) ? 0 : (h_idx == 1) ? 18 : 28;
  const int gstr  = (h_idx == 0) ? 6 : 5;
  float m = -3.402823466e38f;
  for (int p = 0; p < P; ++p) {
    const int g = gbase + p * gstr + j;
    const float* pb = part + (size_t)g * KSPLIT * (S_LEN * CH) + (size_t)s * CH + o;
    float sum = 0.f;
#pragma unroll
    for (int c = 0; c < KSPLIT; ++c) sum += pb[(size_t)c * (S_LEN * CH)];
    m = fmaxf(m, sum);
  }
  const float* bias = (h_idx == 0) ? b1 : (h_idx == 1) ? b2 : b3;
  const float v = m + bias[j * CH + o];
  out[(size_t)o * (S_LEN * 16) + (size_t)s * 16 + kidx] = fmaxf(v, 0.f);
}

// ================= FALLBACK PATH (small ws): direct fp32 -> cvt -> LDS =================
#define D_MT 64
#define D_NT 64
#define D_KB 32
#define D_STRIDE 40

__global__ __launch_bounds__(128)
void vrtconv_gemm_direct(const float* __restrict__ ec,
                         const float* __restrict__ w1,
                         const float* __restrict__ w2,
                         const float* __restrict__ w3,
                         float* __restrict__ conv) {
  __shared__ __bf16 As[2][D_MT][D_STRIDE];
  __shared__ __bf16 Bs[2][D_NT][D_STRIDE];

  const int g = blockIdx.z;
  int h_idx, p, j;
  decode_gemm(g, h_idx, p, j);
  const int h = h_idx + 1;
  const float* __restrict__ w = (h_idx == 0) ? w1 : (h_idx == 1) ? w2 : w3;

  const int mBase = blockIdx.x * D_MT;
  const int nBase = blockIdx.y * D_NT;

  const int tid  = threadIdx.x;
  const int lane = tid & 31;
  const int wv   = tid >> 5;
  const int wm   = (wv >> 1) * 32;
  const int wn   = (wv & 1)  * 32;

  const int lrow = tid & 63;
  const int col0 = (tid >> 6) * 16;

  const int nk = (h * CH * DW) / D_KB;

  const size_t sA = (size_t)(mBase + lrow);
  const size_t oB = (size_t)(nBase + lrow);

  v8f acc[2][2];
#pragma unroll
  for (int a0 = 0; a0 < 2; ++a0)
#pragma unroll
    for (int b0 = 0; b0 < 2; ++b0)
#pragma unroll
      for (int e = 0; e < 8; ++e) acc[a0][b0][e] = 0.f;

  float4 ra[4], rb[4];

  auto addr_pair = [&](int kb, const float** ap, const float** bp) {
    const int k0 = kb * D_KB;
    const int ay = k0 >> 7;
    const int d0 = (k0 & 127) + col0;
    const int a  = ay / h;
    const int y  = ay % h;
    *ap = ec + ((((size_t)a * S_LEN + sA) * 3 + (size_t)(p + y)) << 7) + d0;
    *bp = w  + (((((size_t)(j * CH) + oB) * CH + (size_t)a) * h + (size_t)y) << 7) + d0;
  };

  auto ld_global = [&](int kb) {
    const float *ap, *bp;
    addr_pair(kb, &ap, &bp);
#pragma unroll
    for (int i = 0; i < 4; ++i) ra[i] = ((const float4*)ap)[i];
#pragma unroll
    for (int i = 0; i < 4; ++i) rb[i] = ((const float4*)bp)[i];
    if (kb + 2 < nk) {
      const float *pa, *pb;
      addr_pair(kb + 2, &pa, &pb);
      __builtin_prefetch(pa, 0, 3);
      __builtin_prefetch(pb, 0, 3);
    }
  };

  auto st_lds = [&](int buf) {
    __bf16* da = &As[buf][lrow][col0];
    __bf16* db = &Bs[buf][lrow][col0];
    *(v8bf*)(da)     = cvt8(ra[0], ra[1]);
    *(v8bf*)(da + 8) = cvt8(ra[2], ra[3]);
    *(v8bf*)(db)     = cvt8(rb[0], rb[1]);
    *(v8bf*)(db + 8) = cvt8(rb[2], rb[3]);
  };

  auto compute = [&](int buf) {
    const __bf16* ab = &As[buf][0][0];
    const __bf16* bb = &Bs[buf][0][0];
    v16bf af[2], bfr[2];
#pragma unroll
    for (int mt = 0; mt < 2; ++mt) af[mt]  = load_frag_a<D_STRIDE>(ab, wm + mt * 16, 0, lane);
#pragma unroll
    for (int nt = 0; nt < 2; ++nt) bfr[nt] = load_frag_b<D_STRIDE>(bb, wn + nt * 16, 0, lane);
#pragma unroll
    for (int mt = 0; mt < 2; ++mt)
#pragma unroll
      for (int nt = 0; nt < 2; ++nt)
        acc[mt][nt] = __builtin_amdgcn_wmma_f32_16x16x32_bf16(
            false, af[mt], false, bfr[nt], (short)0, acc[mt][nt], false, false);
  };

  ld_global(0);
  st_lds(0);
  __syncthreads();
  for (int kb = 0; kb < nk; ++kb) {
    const int  cur  = kb & 1;
    const bool more = (kb + 1) < nk;
    if (more) ld_global(kb + 1);
    compute(cur);
    if (more) st_lds(cur ^ 1);
    __syncthreads();
  }

  float* cbase = conv + (size_t)g * (S_LEN * CH);
  const int mrow = (lane >> 4) * 8;
  const int ncol = lane & 15;
#pragma unroll
  for (int mt = 0; mt < 2; ++mt)
#pragma unroll
    for (int nt = 0; nt < 2; ++nt) {
      const int sOut = mBase + wm + mt * 16 + mrow;
      const int oOut = nBase + wn + nt * 16 + ncol;
#pragma unroll
      for (int r = 0; r < 8; ++r)
        cbase[(size_t)(sOut + r) * CH + oOut] = acc[mt][nt][r];
    }
}

__global__ __launch_bounds__(256)
void vrtconv_reduce_direct(const float* __restrict__ conv,
                           const float* __restrict__ b1,
                           const float* __restrict__ b2,
                           const float* __restrict__ b3,
                           float* __restrict__ out) {
  const int t = blockIdx.x * blockDim.x + threadIdx.x;
  if (t >= CH * S_LEN * 16) return;
  const int kidx = t & 15;
  const int s    = (t >> 4) & (S_LEN - 1);
  const int o    = t >> 12;
  const int h_idx = kidx % 3;
  const int j     = kidx / 3;
  const int P     = 3 - h_idx;
  const int gbase = (h_idx == 0) ? 0 : (h_idx == 1) ? 18 : 28;
  const int gstr  = (h_idx == 0) ? 6 : 5;
  float m = -3.402823466e38f;
  for (int p = 0; p < P; ++p)
    m = fmaxf(m, conv[(size_t)(gbase + p * gstr + j) * (S_LEN * CH) + (size_t)s * CH + o]);
  const float* bias = (h_idx == 0) ? b1 : (h_idx == 1) ? b2 : b3;
  const float v = m + bias[j * CH + o];
  out[(size_t)o * (S_LEN * 16) + (size_t)s * 16 + kidx] = fmaxf(v, 0.f);
}

// ================= launch =================
extern "C" void kernel_launch(void* const* d_in, const int* in_sizes, int n_in,
                              void* d_out, int out_size, void* d_ws, size_t ws_size,
                              hipStream_t stream) {
  const float* ec = (const float*)d_in[0];
  const float* w1 = (const float*)d_in[1];
  const float* w2 = (const float*)d_in[2];
  const float* w3 = (const float*)d_in[3];
  const float* b1 = (const float*)d_in[4];
  const float* b2 = (const float*)d_in[5];
  const float* b3 = (const float*)d_in[6];
  float* out = (float*)d_out;

  const size_t nEc = (size_t)CH * S_LEN * 3 * DW;        // 12.58M
  const size_t nW1 = (size_t)6 * CH * CH * 1 * DW;       // 12.58M
  const size_t nW2 = (size_t)5 * CH * CH * 2 * DW;       // 20.97M
  const size_t nW3 = (size_t)5 * CH * CH * 3 * DW;       // 31.46M
  const size_t offW1 = nEc * 2;
  const size_t offW2 = offW1 + nW1 * 2;
  const size_t offW3 = offW2 + nW2 * 2;
  const size_t offP  = offW3 + nW3 * 2;
  const size_t partBytes = (size_t)NGEMM * KSPLIT * S_LEN * CH * sizeof(float);
  const size_t needFast = offP + partBytes;
  const int total = CH * S_LEN * 16;

  if (ws_size >= needFast) {
    char* ws = (char*)d_ws;
    __bf16* ecb = (__bf16*)ws;
    __bf16* w1b = (__bf16*)(ws + offW1);
    __bf16* w2b = (__bf16*)(ws + offW2);
    __bf16* w3b = (__bf16*)(ws + offW3);
    float*  part = (float*)(ws + offP);

    cvt_f32_bf16<<<(int)(nEc / 8 / 256), 256, 0, stream>>>(ec, ecb, (int)(nEc / 8));
    cvt_f32_bf16<<<(int)(nW1 / 8 / 256), 256, 0, stream>>>(w1, w1b, (int)(nW1 / 8));
    cvt_f32_bf16<<<(int)(nW2 / 8 / 256), 256, 0, stream>>>(w2, w2b, (int)(nW2 / 8));
    cvt_f32_bf16<<<(int)(nW3 / 8 / 256), 256, 0, stream>>>(w3, w3b, (int)(nW3 / 8));

    dim3 grid(S_LEN / F_MT, KSPLIT, NGEMM);   // (2, 8, 33) = 528 blocks x 8 waves
    vrtconv_gemm_async<<<grid, 256, 0, stream>>>(ecb, w1b, w2b, w3b, part);
    vrtconv_reduce_split<<<(total + 255) / 256, 256, 0, stream>>>(part, b1, b2, b3, out);
  } else {
    float* conv = (float*)d_ws;               // 4.33 MB
    dim3 grid(S_LEN / D_MT, CH / D_NT, NGEMM);
    vrtconv_gemm_direct<<<grid, 128, 0, stream>>>(ec, w1, w2, w3, conv);
    vrtconv_reduce_direct<<<(total + 255) / 256, 256, 0, stream>>>(conv, b1, b2, b3, out);
  }
}
